// MaskedSelfAttention_74586402062627
// MI455X (gfx1250) — compile-verified, matrix operand order checked by reading, not run on hardware
//
#include <hip/hip_runtime.h>

typedef __bf16 bf16_t;
typedef __attribute__((ext_vector_type(16))) __bf16 v16bf;
typedef __attribute__((ext_vector_type(8)))  __bf16 v8bf;
typedef __attribute__((ext_vector_type(8)))  float   v8f;

#define BATCH 32
#define SEQ   1000
#define EMB   1024
#define ADIM  512
#define SPAD  1024   // S padded to 64 tiles of 16 (so 32-key steps never go OOB)

__device__ __forceinline__ v16bf cat8(v8bf lo, v8bf hi) {
  return __builtin_shufflevector(lo, hi, 0,1,2,3,4,5,6,7,8,9,10,11,12,13,14,15);
}

__device__ __forceinline__ v8f wmma_bf16(v16bf a, v16bf b, v8f c) {
  return __builtin_amdgcn_wmma_f32_16x16x32_bf16(false, a, false, b, (short)0, c,
                                                 false, false);
}

// ---------------------------------------------------------------------------
// Kernel 0: transpose + convert one weight matrix W[E,A] f32 -> Wt[A,E] bf16
// ---------------------------------------------------------------------------
__global__ __launch_bounds__(256) void wt_kernel(const float* __restrict__ W,
                                                 bf16_t* __restrict__ Wt) {
  const int idx = blockIdx.x * 256 + threadIdx.x;   // 524288 elements
  const int e = idx & (EMB - 1);
  const int a = idx >> 10;
  Wt[(size_t)a * EMB + e] = (bf16_t)W[(size_t)e * ADIM + a];
}

// ---------------------------------------------------------------------------
// Kernel 1: projection  out = x @ W + bias   (one launch per weight)
//   x:[B,S,E] f32, Wt:[A,E] bf16 (pre-transposed), bias:[A] f32
//   Each wave: one 16-row M tile x four 16-col N tiles; K loop of 32.
//   transpose_out==0 : out[B*SPAD, A]   (Q, K)
//   transpose_out==1 : out[B*A, SPAD]   (V^T)
// ---------------------------------------------------------------------------
__global__ __launch_bounds__(256) void qkv_proj_kernel(
    const float* __restrict__ x, const bf16_t* __restrict__ Wt,
    const float* __restrict__ bias, bf16_t* __restrict__ out, int transpose_out)
{
  const int tid  = threadIdx.x;
  const int wave = tid >> 5;
  const int lane = tid & 31;
  const int half = lane >> 4;
  const int ln   = lane & 15;

  const int grp = blockIdx.x * 8 + wave;    // 16384 groups
  const int tm  = grp >> 3;                 // 2048 M tiles (B*SPAD/16)
  const int ng  = grp & 7;                  // 8 N groups of 4 tiles

  const int m0    = tm * 16;
  const int row   = m0 + ln;                // padded row
  const int batch = row >> 10;
  const int s     = row & (SPAD - 1);
  const int sc    = (s < SEQ) ? s : (SEQ - 1);  // clamp: loads always in-bounds,
                                                // pad rows are zeroed at store time
  const float* xrow = x + ((size_t)batch * SEQ + sc) * EMB;

  v8f acc[4] = {};
  for (int ke = 0; ke < EMB; ke += 32) {
    // A fragment: two contiguous 8-float runs -> 4x b128 loads, cvt to bf16
    const float4 a0 = *reinterpret_cast<const float4*>(xrow + ke + half * 8);
    const float4 a1 = *reinterpret_cast<const float4*>(xrow + ke + half * 8 + 4);
    const float4 a2 = *reinterpret_cast<const float4*>(xrow + ke + 16 + half * 8);
    const float4 a3 = *reinterpret_cast<const float4*>(xrow + ke + 16 + half * 8 + 4);
    v16bf aM;
    aM[0]  = (bf16_t)a0.x; aM[1]  = (bf16_t)a0.y; aM[2]  = (bf16_t)a0.z; aM[3]  = (bf16_t)a0.w;
    aM[4]  = (bf16_t)a1.x; aM[5]  = (bf16_t)a1.y; aM[6]  = (bf16_t)a1.z; aM[7]  = (bf16_t)a1.w;
    aM[8]  = (bf16_t)a2.x; aM[9]  = (bf16_t)a2.y; aM[10] = (bf16_t)a2.z; aM[11] = (bf16_t)a2.w;
    aM[12] = (bf16_t)a3.x; aM[13] = (bf16_t)a3.y; aM[14] = (bf16_t)a3.z; aM[15] = (bf16_t)a3.w;

#pragma unroll
    for (int t = 0; t < 4; ++t) {
      const int col = (ng * 4 + t) * 16 + ln;
      const bf16_t* wrow = Wt + (size_t)col * EMB + ke + half * 8;
      v16bf bM = cat8(*reinterpret_cast<const v8bf*>(wrow),
                      *reinterpret_cast<const v8bf*>(wrow + 16));
      acc[t] = wmma_bf16(aM, bM, acc[t]);
    }
  }

#pragma unroll
  for (int t = 0; t < 4; ++t) {
    const int col = (ng * 4 + t) * 16 + ln;
    const float bv = bias[col];
#pragma unroll
    for (int r = 0; r < 8; ++r) {
      const int M    = r + 8 * half;
      const int prow = m0 + M;
      const int prs  = prow & (SPAD - 1);
      const float val = (prs < SEQ) ? (acc[t][r] + bv) : 0.0f;  // zero pad rows
      if (!transpose_out) {
        out[(size_t)prow * ADIM + col] = (bf16_t)val;
      } else {
        const int pb = prow >> 10;
        out[((size_t)pb * ADIM + col) * SPAD + prs] = (bf16_t)val;
      }
    }
  }
}

// ---------------------------------------------------------------------------
// Kernel 2: causal flash attention
//   grid = (63 query tiles, 32 batches), block = 64 (2 waves)
//   wave w covers output columns [w*256, w*256+256)
// ---------------------------------------------------------------------------
__global__ __launch_bounds__(64) void attn_kernel(
    const bf16_t* __restrict__ qb,   // [B*SPAD, ADIM]
    const bf16_t* __restrict__ kb,   // [B*SPAD, ADIM]
    const bf16_t* __restrict__ vt,   // [B*ADIM, SPAD]  (V transposed)
    float* __restrict__ out)         // [B, SEQ, ADIM]
{
  __shared__ bf16_t qs[16 * ADIM];        // 16 KB: the query tile
  __shared__ bf16_t ps[2][16 * 32];       // per-wave P tile (D-layout -> A-layout relay)

  const int qt   = blockIdx.x;            // 0..62
  const int b    = blockIdx.y;            // 0..31
  const int tid  = threadIdx.x;
  const int wave = tid >> 5;
  const int lane = tid & 31;
  const int half = lane >> 4;
  const int ln   = lane & 15;
  const int cbase = wave * 256;

  // cooperative Q tile load into LDS (16 rows x 512 bf16 = 1024 int4 chunks)
  {
    const int4* src = reinterpret_cast<const int4*>(qb + ((size_t)b * SPAD + qt * 16) * ADIM);
    int4* dst = reinterpret_cast<int4*>(qs);
#pragma unroll
    for (int i = 0; i < 16; ++i) dst[tid + i * 64] = src[tid + i * 64];
  }
  __syncthreads();

  v8f acc[16] = {};
  float mrow[8], lrow[8];
#pragma unroll
  for (int r = 0; r < 8; ++r) { mrow[r] = -3.0e38f; lrow[r] = 0.0f; }

  const float scale = 0.031622776601683794f;   // 1/sqrt(1000)
  const int npairs = (qt + 2) >> 1;            // ceil((qt+1)/2) 32-key steps

  for (int kp = 0; kp < npairs; ++kp) {
    const int k0 = kp * 32;
    const bf16_t* krow = kb + ((size_t)b * SPAD + k0) * ADIM;

    // ---- scores: S[16q x 32k] = Q(16x512) . K^T(512x32), 32 WMMAs ----
    v8f s0 = {}, s1 = {};
    for (int kk = 0; kk < 16; ++kk) {
      const int d0 = kk * 32 + half * 8;
      v16bf aQ = cat8(*reinterpret_cast<const v8bf*>(&qs[ln * ADIM + d0]),
                      *reinterpret_cast<const v8bf*>(&qs[ln * ADIM + d0 + 16]));
      v16bf bK0 = cat8(*reinterpret_cast<const v8bf*>(&krow[(size_t)ln * ADIM + d0]),
                       *reinterpret_cast<const v8bf*>(&krow[(size_t)ln * ADIM + d0 + 16]));
      v16bf bK1 = cat8(*reinterpret_cast<const v8bf*>(&krow[(size_t)(16 + ln) * ADIM + d0]),
                       *reinterpret_cast<const v8bf*>(&krow[(size_t)(16 + ln) * ADIM + d0 + 16]));
      s0 = wmma_bf16(aQ, bK0, s0);
      s1 = wmma_bf16(aQ, bK1, s1);
    }

    // ---- online softmax (row M = r + 8*half lives in one 16-lane half) ----
#pragma unroll
    for (int r = 0; r < 8; ++r) {
      const int M  = r + 8 * half;
      const int i  = qt * 16 + M;
      const int j0 = k0 + ln;
      const int j1 = k0 + 16 + ln;
      float v0 = (j0 <= i) ? s0[r] * scale : -3.0e38f;
      float v1 = (j1 <= i) ? s1[r] * scale : -3.0e38f;
      float mx = fmaxf(v0, v1);
      mx = fmaxf(mx, __shfl_xor(mx, 1, 16));
      mx = fmaxf(mx, __shfl_xor(mx, 2, 16));
      mx = fmaxf(mx, __shfl_xor(mx, 4, 16));
      mx = fmaxf(mx, __shfl_xor(mx, 8, 16));
      const float mprev = mrow[r];
      const float mcur  = fmaxf(mprev, mx);
      const float alpha = __expf(mprev - mcur);
      const float p0 = __expf(v0 - mcur);
      const float p1 = __expf(v1 - mcur);
      float sum = p0 + p1;
      sum += __shfl_xor(sum, 1, 16);
      sum += __shfl_xor(sum, 2, 16);
      sum += __shfl_xor(sum, 4, 16);
      sum += __shfl_xor(sum, 8, 16);
      lrow[r] = lrow[r] * alpha + sum;
      mrow[r] = mcur;
#pragma unroll
      for (int nt = 0; nt < 16; ++nt) acc[nt][r] *= alpha;
      ps[wave][M * 32 + ln]      = (bf16_t)p0;
      ps[wave][M * 32 + 16 + ln] = (bf16_t)p1;
    }
    __syncthreads();   // order LDS P writes before reads (uniform loop count)

    // ---- O += P(16x32) . V(32x16cols) per N tile, 16 WMMAs ----
    v16bf aP = cat8(*reinterpret_cast<const v8bf*>(&ps[wave][ln * 32 + half * 8]),
                    *reinterpret_cast<const v8bf*>(&ps[wave][ln * 32 + 16 + half * 8]));
#pragma unroll
    for (int nt = 0; nt < 16; ++nt) {
      const int c = cbase + nt * 16 + ln;
      const bf16_t* vcol = vt + ((size_t)b * ADIM + c) * SPAD + k0 + half * 8;
      v16bf bV = cat8(*reinterpret_cast<const v8bf*>(vcol),
                      *reinterpret_cast<const v8bf*>(vcol + 16));
      acc[nt] = wmma_bf16(aP, bV, acc[nt]);
    }
    __syncthreads();
  }

  // ---- normalize and store valid rows ----
#pragma unroll
  for (int r = 0; r < 8; ++r) {
    const int M = r + 8 * half;
    const int s = qt * 16 + M;
    if (s < SEQ) {
      const float inv = 1.0f / lrow[r];
#pragma unroll
      for (int nt = 0; nt < 16; ++nt) {
        const int c = cbase + nt * 16 + ln;
        out[((size_t)b * SEQ + s) * ADIM + c] = acc[nt][r] * inv;
      }
    }
  }
}

// ---------------------------------------------------------------------------
extern "C" void kernel_launch(void* const* d_in, const int* in_sizes, int n_in,
                              void* d_out, int out_size, void* d_ws, size_t ws_size,
                              hipStream_t stream) {
  (void)in_sizes; (void)n_in; (void)out_size; (void)ws_size;
  const float* x  = (const float*)d_in[0];
  const float* Wk = (const float*)d_in[1];
  const float* bk = (const float*)d_in[2];
  const float* Wq = (const float*)d_in[3];
  const float* bq = (const float*)d_in[4];
  const float* Wv = (const float*)d_in[5];
  const float* bv = (const float*)d_in[6];

  const size_t plane = (size_t)BATCH * SPAD * ADIM;   // 16.78M bf16 elems each
  const size_t wplane = (size_t)EMB * ADIM;           // 0.52M bf16 elems each
  bf16_t* qb  = (bf16_t*)d_ws;                        // ws: 3*plane*2 + 3*wplane*2 ~ 104 MB
  bf16_t* kb  = qb + plane;
  bf16_t* vt  = kb + plane;
  bf16_t* wtq = vt + plane;
  bf16_t* wtk = wtq + wplane;
  bf16_t* wtv = wtk + wplane;

  // weight transpose/convert (tiny)
  wt_kernel<<<dim3(2048), dim3(256), 0, stream>>>(Wq, wtq);
  wt_kernel<<<dim3(2048), dim3(256), 0, stream>>>(Wk, wtk);
  wt_kernel<<<dim3(2048), dim3(256), 0, stream>>>(Wv, wtv);

  dim3 pblk(256), pgrd(2048);                          // 2048*8 waves = 16384 tile groups
  qkv_proj_kernel<<<pgrd, pblk, 0, stream>>>(x, wtq, bq, qb, 0);
  qkv_proj_kernel<<<pgrd, pblk, 0, stream>>>(x, wtk, bk, kb, 0);
  qkv_proj_kernel<<<pgrd, pblk, 0, stream>>>(x, wtv, bv, vt, 1);

  attn_kernel<<<dim3(63, 32), dim3(64), 0, stream>>>(qb, kb, vt, (float*)d_out);
}
